// MultiScaleVQVAE_69569880261356
// MI455X (gfx1250) — matrix operand order repaired and checked by reading, not compiled, not run on hardware
//
#include <hip/hip_runtime.h>
#include <hip/hip_bf16.h>
#include <math.h>

// ---------------------------------------------------------------------------
// MultiScaleVQVAE forward for MI455X (gfx1250, wave32, WMMA).
// HBM-bound workload (23.3 TB/s roof): activations stored f16 (halves traffic),
// all GEMMs via v_wmma_f32_16x16x32_f16 with f32 accumulation, fragments read
// from LDS as ds_load_b128 vectors, async global->LDS DMA (ASYNCcnt) for the
// VQ codebook stream and the K-contiguous attention GEMM tiles.
// ---------------------------------------------------------------------------

typedef _Float16 half_t;
typedef __attribute__((ext_vector_type(16))) _Float16 v16h;
typedef __attribute__((ext_vector_type(8)))  _Float16 v8h;
typedef __attribute__((ext_vector_type(2)))  _Float16 v2h;
typedef __attribute__((ext_vector_type(8)))  float    v8f;

#define BLK 128            // 4 waves per workgroup
#define TM 16
#define TK 32
#define TN 64              // 4 waves x 16 cols
#define LDP 40             // padded LDS row stride in halves (80B: 16B aligned, bank-skewed)

// A fragment (16-bit A 16x32, ISA 7.12.2): per lane-half h, the 16 K values are
// two contiguous runs: [h*8, h*8+8) and [16+h*8, 16+h*8+8)  -> two b128 loads.
__device__ __forceinline__ v16h loadA(const half_t* row, int h) {
    v8h a0 = *(const v8h*)(row + h * 8);
    v8h a1 = *(const v8h*)(row + 16 + h * 8);
    return __builtin_shufflevector(a0, a1, 0, 1, 2, 3, 4, 5, 6, 7,
                                   8, 9, 10, 11, 12, 13, 14, 15);
}
// B fragment (16-bit B 32x16): per lane-half h, same K order as A: runs
// [h*8, h*8+8) and [16+h*8, 16+h*8+8) of the column-major tile (one LDS row per column).
__device__ __forceinline__ v16h loadB(const half_t* col, int h) {
    v8h b0 = *(const v8h*)(col + h * 8);
    v8h b1 = *(const v8h*)(col + 16 + h * 8);
    return __builtin_shufflevector(b0, b1, 0, 1, 2, 3, 4, 5, 6, 7,
                                   8, 9, 10, 11, 12, 13, 14, 15);
}
// async DMA of one 16B chunk: uniform SGPR base + per-lane LDS/global offsets
__device__ __forceinline__ void asyncB128(unsigned ldsAddr, unsigned gOff, const void* base) {
    asm volatile("global_load_async_to_lds_b128 %0, %1, %2"
                 :: "v"(ldsAddr), "v"(gOff), "s"(base) : "memory");
}

// ---------------------------------------------------------------------------
// Implicit-GEMM conv1d (f16 in / f16 or f32 out, f32 weights converted while
// staging, f32 accumulate).  KW is compile-time so im2col div/mod folds away.
// Out = alpha*(conv + bias) + beta*Res.  Arbitrary strides: NCH<->NHC
// transposes and ConvTranspose even/odd interleave are free.
// ---------------------------------------------------------------------------
template <int KW>
__global__ void k_conv1d_wmma(const half_t* __restrict__ X, long xbs, int xcs, int xps,
                              int Cin, int Lin,
                              const float* __restrict__ W, const float* __restrict__ Bi,
                              int strd, int padL,
                              const half_t* __restrict__ Res, float alpha, float beta,
                              half_t* __restrict__ Out, float* __restrict__ OutF32,
                              long obs, int ocs, int ops, long ooff,
                              int Cout, int Lout, int tilesM)
{
    __shared__ half_t sA[TM][LDP];   // [m][k]
    __shared__ half_t sB[TN][LDP];   // [n][k]  (column-major tile)

    const int b    = blockIdx.x / tilesM;
    const int m0   = (blockIdx.x % tilesM) * TM;
    const int n0b  = blockIdx.y * TN;
    const int tid  = threadIdx.x;
    const int wave = tid >> 5;
    const int lane = tid & 31;
    const int Ktot = Cin * KW;
    const half_t* xb = X + (long)b * xbs;

    v8f acc = {};

    for (int k0 = 0; k0 < Ktot; k0 += TK) {
        // stage A tile (im2col gather), paired b32 stores; KW-div is compile-time
        for (int i = tid; i < TM * TK / 2; i += BLK) {
            int m  = i >> 4;
            int kk = (i & 15) * 2;
            half_t v0 = (half_t)0.f, v1 = (half_t)0.f;
            int kg = k0 + kk;
            if (kg < Ktot) {
                int c = kg / KW, t = kg - c * KW;
                int ip = (m0 + m) * strd + t - padL;
                if (ip >= 0 && ip < Lin) v0 = xb[(long)c * xcs + (long)ip * xps];
            }
            if (kg + 1 < Ktot) {
                int c = (kg + 1) / KW, t = (kg + 1) - c * KW;
                int ip = (m0 + m) * strd + t - padL;
                if (ip >= 0 && ip < Lin) v1 = xb[(long)c * xcs + (long)ip * xps];
            }
            v2h pk = {v0, v1};
            *(v2h*)&sA[m][kk] = pk;
        }
        // stage B tile (weights f32 -> f16), column-major, paired stores
        for (int i = tid; i < TN * TK / 2; i += BLK) {
            int n  = i >> 4;
            int kk = (i & 15) * 2;
            int co = n0b + n;
            float v0 = 0.f, v1 = 0.f;
            if (co < Cout) {
                int kg = k0 + kk;
                if (kg < Ktot) {
                    int c = kg / KW, t = kg - c * KW;
                    v0 = W[((long)co * Cin + c) * KW + t];
                }
                if (kg + 1 < Ktot) {
                    int c = (kg + 1) / KW, t = (kg + 1) - c * KW;
                    v1 = W[((long)co * Cin + c) * KW + t];
                }
            }
            v2h pk = {(half_t)v0, (half_t)v1};
            *(v2h*)&sB[n][kk] = pk;
        }
        if (tid == 0 && k0 + TK < Ktot)
            __builtin_prefetch(&W[((long)n0b * Cin) * KW + k0 + TK], 0, 1);
        __syncthreads();

        const int ml = lane & 15, h = lane >> 4;
        v16h af = loadA(&sA[ml][0], h);
        v16h bf = loadB(&sB[wave * 16 + ml][0], h);
        acc = __builtin_amdgcn_wmma_f32_16x16x32_f16(false, af, false, bf,
                                                     (short)0, acc, false, false);
        __syncthreads();
    }

    const int n = n0b + wave * 16 + (lane & 15);
    if (n < Cout) {
        const float bi = Bi ? Bi[n] : 0.f;
#pragma unroll
        for (int r = 0; r < 8; ++r) {
            int m = m0 + r + ((lane >> 4) << 3);
            long o = (long)b * obs + (long)n * ocs + (long)m * ops + ooff;
            float v = alpha * (acc[r] + bi);
            if (beta != 0.f && Res) v += beta * (float)Res[o];
            if (OutF32) OutF32[o] = v;
            else        Out[o]    = (half_t)v;
        }
    }
}

// ---------------------------------------------------------------------------
// Strided batched GEMM (f16 in/out, f32 accumulate): C[b] = alpha*A[b]*B[b].
// ASYNCP=true: tiles are K-contiguous (acs==1 && brs==1, full tiles) and are
// filled by global_load_async_to_lds_b128 DMA instead of VGPR round-trips.
// ---------------------------------------------------------------------------
template <bool ASYNCP>
__global__ void k_bgemm_wmma(const half_t* __restrict__ A, long abss, int ars, int acs,
                             const half_t* __restrict__ Bm, long bbs, int brs, int bcs,
                             half_t* __restrict__ Cc, long cbs, int crs, int ccs,
                             int M, int N, int Kd, float alpha, int tilesM)
{
    __shared__ half_t sA[TM][LDP];
    __shared__ half_t sB[TN][LDP];

    const int bi   = blockIdx.x / tilesM;
    const int m0   = (blockIdx.x % tilesM) * TM;
    const int n0b  = blockIdx.y * TN;
    const int tid  = threadIdx.x;
    const int wave = tid >> 5;
    const int lane = tid & 31;
    const half_t* Ab = A + (long)bi * abss;
    const half_t* Bb = Bm + (long)bi * bbs;

    v8f acc = {};

    for (int k0 = 0; k0 < Kd; k0 += TK) {
        if constexpr (ASYNCP) {
            // A: 16 rows x 64B = 64 chunks; B: 64 cols x 64B = 256 chunks
            for (int ch = tid; ch < 64 + 256; ch += BLK) {
                if (ch < 64) {
                    int m = ch >> 2, part = ch & 3;
                    unsigned lds = (unsigned)(size_t)&sA[m][part * 8];
                    unsigned go  = (unsigned)(((long)(m0 + m) * ars + k0 + part * 8) * 2);
                    asyncB128(lds, go, Ab);
                } else {
                    int c2 = ch - 64;
                    int n = c2 >> 2, part = c2 & 3;
                    unsigned lds = (unsigned)(size_t)&sB[n][part * 8];
                    unsigned go  = (unsigned)(((long)(n0b + n) * bcs + k0 + part * 8) * 2);
                    asyncB128(lds, go, Bb);
                }
            }
            asm volatile("s_wait_asynccnt 0" ::: "memory");
        } else {
            for (int i = tid; i < TM * TK / 2; i += BLK) {
                int m = i >> 4, kk = (i & 15) * 2;
                int kg = k0 + kk;
                half_t v0 = (kg     < Kd) ? Ab[(long)(m0 + m) * ars + (long)kg * acs]       : (half_t)0.f;
                half_t v1 = (kg + 1 < Kd) ? Ab[(long)(m0 + m) * ars + (long)(kg + 1) * acs] : (half_t)0.f;
                v2h pk = {v0, v1};
                *(v2h*)&sA[m][kk] = pk;
            }
            for (int i = tid; i < TN * TK / 2; i += BLK) {
                int n = i >> 4, kk = (i & 15) * 2;
                int kg = k0 + kk, nc = n0b + n;
                half_t v0 = (half_t)0.f, v1 = (half_t)0.f;
                if (nc < N) {
                    if (kg     < Kd) v0 = Bb[(long)kg * brs + (long)nc * bcs];
                    if (kg + 1 < Kd) v1 = Bb[(long)(kg + 1) * brs + (long)nc * bcs];
                }
                v2h pk = {v0, v1};
                *(v2h*)&sB[n][kk] = pk;
            }
        }
        __syncthreads();

        const int ml = lane & 15, h = lane >> 4;
        v16h af = loadA(&sA[ml][0], h);
        v16h bf = loadB(&sB[wave * 16 + ml][0], h);
        acc = __builtin_amdgcn_wmma_f32_16x16x32_f16(false, af, false, bf,
                                                     (short)0, acc, false, false);
        __syncthreads();
    }

    const int n = n0b + wave * 16 + (lane & 15);
    if (n < N) {
        half_t* Cb = Cc + (long)bi * cbs;
#pragma unroll
        for (int r = 0; r < 8; ++r) {
            int m = m0 + r + ((lane >> 4) << 3);
            Cb[(long)m * crs + (long)n * ccs] = (half_t)(alpha * acc[r]);
        }
    }
}

// ---------------------------------------------------------------------------
// Fused VQ nearest-codebook search.  score = -2*z.cb^T + |cb|^2 (|z|^2 is
// row-constant, dropped).  K = ZCH = 32 -> one WMMA per 16x16 tile.  The f16
// codebook is streamed through LDS with double-buffered async DMA
// (global_load_async_to_lds_b128 + s_wait_asynccnt), one wave per block.
// ---------------------------------------------------------------------------
__global__ void k_vq_argmin_wmma(const half_t* __restrict__ z, int pn, int C,
                                 const half_t* __restrict__ cbh,
                                 const float* __restrict__ e2, int V,
                                 int* __restrict__ idx)
{
    __shared__ half_t scb[2][16][LDP];   // 16 codebook rows x 32 halves, padded

    const int r0   = blockIdx.x * 16;
    const int lane = threadIdx.x;
    const int ml   = lane & 15, h = lane >> 4;

    // A fragment: row r0+ml, two contiguous 8-half K runs
    v16h af;
    {
        int r = r0 + ml;
        int b = r / pn, p = r - b * pn;
#pragma unroll
        for (int j = 0; j < 8; ++j)
            af[j] = z[((long)b * C + (h * 8 + j)) * pn + p];
#pragma unroll
        for (int j = 0; j < 8; ++j)
            af[8 + j] = z[((long)b * C + (16 + h * 8 + j)) * pn + p];
    }

    float best[8]; int bidx[8];
#pragma unroll
    for (int r = 0; r < 8; ++r) { best[r] = 3.4e38f; bidx[r] = 0; }

    const int T = V / 16;

    // each lane DMAs two 16B chunks per 16-row tile (16 rows x 64B = 64 chunks)
    auto issue = [&](int tile, int buf) {
#pragma unroll
        for (int q = 0; q < 2; ++q) {
            int chunk = lane + q * 32;
            int row = chunk >> 2, part = chunk & 3;
            unsigned lds = (unsigned)(size_t)&scb[buf][row][part * 8];
            unsigned go  = (unsigned)(((tile * 16 + row) * C + part * 8) * 2);
            asyncB128(lds, go, cbh);
        }
    };
    issue(0, 0);
    for (int t = 0; t < T; ++t) {
        const int buf = t & 1;
        if (t + 1 < T) {
            issue(t + 1, buf ^ 1);
            asm volatile("s_wait_asynccnt 2" ::: "memory");
        } else {
            asm volatile("s_wait_asynccnt 0" ::: "memory");
        }
        v16h bf = loadB(&scb[buf][ml][0], h);
        v8f s = {};
        s = __builtin_amdgcn_wmma_f32_16x16x32_f16(false, af, false, bf,
                                                   (short)0, s, false, false);
        const int n = t * 16 + ml;
        const float e = e2[n];
#pragma unroll
        for (int r = 0; r < 8; ++r) {
            float d = e - 2.f * s[r];
            if (d < best[r]) { best[r] = d; bidx[r] = n; }
        }
    }

    __shared__ float sv[16][17];
    __shared__ int   si[16][17];
#pragma unroll
    for (int r = 0; r < 8; ++r) { sv[r + 8 * h][ml] = best[r]; si[r + 8 * h][ml] = bidx[r]; }
    __syncthreads();
    if (lane < 16) {
        float bv = sv[lane][0]; int bi2 = si[lane][0];
        for (int t = 1; t < 16; ++t)
            if (sv[lane][t] < bv) { bv = sv[lane][t]; bi2 = si[lane][t]; }
        idx[r0 + lane] = bi2;
    }
}

// ---------------------------------------------------------------------------
// GroupNorm (f32 statistics over f16 data) + optional fused SiLU.
// ---------------------------------------------------------------------------
__global__ void k_gn(const half_t* __restrict__ X, const float* __restrict__ gamma,
                     const float* __restrict__ beta, half_t* __restrict__ Out,
                     int C, int L, int G, int doSilu)
{
    const int b = blockIdx.x / G, g = blockIdx.x % G;
    const int cpg = C / G;
    const long n = (long)cpg * L;
    const half_t* xp = X + ((long)b * C + (long)g * cpg) * L;
    half_t* op = Out + ((long)b * C + (long)g * cpg) * L;

    float s = 0.f, s2 = 0.f;
    for (long i = threadIdx.x; i < n; i += blockDim.x) {
        float v = (float)xp[i]; s += v; s2 += v * v;
    }
    __shared__ float rs[256], rq[256];
    rs[threadIdx.x] = s; rq[threadIdx.x] = s2;
    __syncthreads();
    for (int st = 128; st > 0; st >>= 1) {
        if (threadIdx.x < st) {
            rs[threadIdx.x] += rs[threadIdx.x + st];
            rq[threadIdx.x] += rq[threadIdx.x + st];
        }
        __syncthreads();
    }
    const float mean = rs[0] / (float)n;
    const float var  = rq[0] / (float)n - mean * mean;
    const float inv  = rsqrtf(var + 1e-6f);
    for (long i = threadIdx.x; i < n; i += blockDim.x) {
        int c = g * cpg + (int)(i / L);
        float v = ((float)xp[i] - mean) * inv * gamma[c] + beta[c];
        if (doSilu) v = v / (1.f + __expf(-v));
        op[i] = (half_t)v;
    }
}

// softmax over last axis (one block per (b, channel) row)
__global__ void k_softmax(const half_t* __restrict__ Xb, long xbs,
                          half_t* __restrict__ Ob, long obs2, int C, int L)
{
    const int rid = blockIdx.x;
    const int b = rid / C, d = rid % C;
    const half_t* x = Xb + (long)b * xbs + (long)d * L;
    half_t* o = Ob + (long)b * obs2 + (long)d * L;
    __shared__ float red[128];
    float mx = -3.4e38f;
    for (int i = threadIdx.x; i < L; i += 128) mx = fmaxf(mx, (float)x[i]);
    red[threadIdx.x] = mx; __syncthreads();
    for (int st = 64; st > 0; st >>= 1) {
        if (threadIdx.x < st) red[threadIdx.x] = fmaxf(red[threadIdx.x], red[threadIdx.x + st]);
        __syncthreads();
    }
    mx = red[0]; __syncthreads();
    float sm = 0.f;
    for (int i = threadIdx.x; i < L; i += 128) sm += __expf((float)x[i] - mx);
    red[threadIdx.x] = sm; __syncthreads();
    for (int st = 64; st > 0; st >>= 1) {
        if (threadIdx.x < st) red[threadIdx.x] += red[threadIdx.x + st];
        __syncthreads();
    }
    const float inv = 1.f / red[0];
    for (int i = threadIdx.x; i < L; i += 128) o[i] = (half_t)(__expf((float)x[i] - mx) * inv);
}

// ------------------------- small elementwise kernels ------------------------
__global__ void k_f32_to_f16(const float* in, half_t* out, long n) {
    long i = (long)blockIdx.x * blockDim.x + threadIdx.x;
    if (i < n) out[i] = (half_t)in[i];
}
__global__ void k_fill16(half_t* p, float v, long n) {
    long i = (long)blockIdx.x * blockDim.x + threadIdx.x;
    if (i < n) p[i] = (half_t)v;
}
__global__ void k_axpy16(half_t* o, const half_t* x, float a, const half_t* y, float b, long n) {
    long i = (long)blockIdx.x * blockDim.x + threadIdx.x;
    if (i < n) o[i] = (half_t)(a * (float)x[i] + (y ? b * (float)y[i] : 0.f));
}
__global__ void k_pool_mean(const half_t* f, half_t* z, int C, int Lz, int pn, long n) {
    long i = (long)blockIdx.x * blockDim.x + threadIdx.x;
    if (i >= n) return;
    int p = (int)(i % pn); long bc = i / pn;
    int w = Lz / pn;
    const half_t* s = f + bc * Lz + (long)p * w;
    float acc = 0.f;
    for (int j = 0; j < w; ++j) acc += (float)s[j];
    z[i] = (half_t)(acc / (float)w);
}
__global__ void k_resize_lin(const half_t* h, half_t* o, int pn, int Lz, long n) {
    long i = (long)blockIdx.x * blockDim.x + threadIdx.x;
    if (i >= n) return;
    int x = (int)(i % Lz); long bc = i / Lz;
    float scale = (float)pn / (float)Lz;
    float s = ((float)x + 0.5f) * scale - 0.5f;
    float fl = floorf(s); int i0 = (int)fl; float fr = s - fl;
    int ia = i0 < 0 ? 0 : (i0 > pn - 1 ? pn - 1 : i0);
    int j1 = i0 + 1;
    int ib = j1 < 0 ? 0 : (j1 > pn - 1 ? pn - 1 : j1);
    const half_t* src = h + bc * pn;
    o[i] = (half_t)((float)src[ia] * (1.f - fr) + (float)src[ib] * fr);
}
__global__ void k_gather_cb(const float* cb, const int* idx, half_t* h, int C, int pn, long n) {
    long i = (long)blockIdx.x * blockDim.x + threadIdx.x;
    if (i >= n) return;
    int p = (int)(i % pn); long bc = i / pn;
    int c = (int)(bc % C); long b = bc / C;
    h[i] = (half_t)cb[(long)idx[b * pn + p] * C + c];
}
__global__ void k_prep_cb(const float* cb, half_t* cbh, float* e2, int V, int C) {
    int v = blockIdx.x * blockDim.x + threadIdx.x;
    if (v >= V) return;
    float s = 0.f;
    for (int c = 0; c < C; ++c) {
        float w = cb[(long)v * C + c];
        s += w * w;
        cbh[(long)v * C + c] = (half_t)w;
    }
    e2[v] = s;
}
// ConvTranspose1d(k=4,s=2,p=1) -> two k=2 convs; repack wT[ci][co][4] into w2[co][ci][2]
__global__ void k_pack_wT(const float* wt, float* w2, int C, int parity) {
    int id = blockIdx.x * blockDim.x + threadIdx.x;
    if (id >= C * C) return;
    int co = id / C, ci = id - co * C;
    int kA = parity ? 2 : 3, kB = parity ? 0 : 1;
    w2[((long)co * C + ci) * 2 + 0] = wt[((long)ci * C + co) * 4 + kA];
    w2[((long)co * C + ci) * 2 + 1] = wt[((long)ci * C + co) * 4 + kB];
}

// ===========================================================================
// Host side
// ===========================================================================
namespace {

struct Conv { const float *b, *w; };
struct Gn   { const float *b, *g; };
struct Res  { Conv c1, c2; Gn n1, n2; Conv sc; bool has_sc; };
struct Attn { Gn n; Conv proj, qkv; };

struct PIn {
    void* const* p; int i;
    const float* next() { return (const float*)p[i++]; }
};
Conv rdConv(PIn& s) { Conv c; c.b = s.next(); c.w = s.next(); return c; }
Gn   rdGn(PIn& s)   { Gn g;  g.b = s.next(); g.g = s.next(); return g; }
Res  rdRes(PIn& s, bool sc) {
    Res r; r.c1 = rdConv(s); r.c2 = rdConv(s); r.n1 = rdGn(s); r.n2 = rdGn(s);
    r.has_sc = sc; r.sc = Conv{nullptr, nullptr};
    if (sc) r.sc = rdConv(s);
    return r;
}
Attn rdAttn(PIn& s) { Attn a; a.n = rdGn(s); a.proj = rdConv(s); a.qkv = rdConv(s); return a; }

constexpr int Bn   = 64;
constexpr int NGRP = 8;

void convRun(hipStream_t st, const half_t* X, int Cin, int Lin,
             long xbs, int xcs, int xps,
             Conv w, int Kw, int strd, int padL,
             const half_t* Rres, float alpha, float beta,
             half_t* Out, long obs, int ocs, int ops, long ooff,
             int Cout, int Lout,
             const float* wOverride = nullptr, float* outF32 = nullptr)
{
    int tilesM = Lout / 16;
    dim3 g(Bn * tilesM, (Cout + TN - 1) / TN);
    const float* wp = wOverride ? wOverride : w.w;
    switch (Kw) {
    case 1:
        k_conv1d_wmma<1><<<g, BLK, 0, st>>>(X, xbs, xcs, xps, Cin, Lin, wp, w.b, strd, padL,
                                            Rres, alpha, beta, Out, outF32,
                                            obs, ocs, ops, ooff, Cout, Lout, tilesM);
        break;
    case 2:
        k_conv1d_wmma<2><<<g, BLK, 0, st>>>(X, xbs, xcs, xps, Cin, Lin, wp, w.b, strd, padL,
                                            Rres, alpha, beta, Out, outF32,
                                            obs, ocs, ops, ooff, Cout, Lout, tilesM);
        break;
    default:
        k_conv1d_wmma<3><<<g, BLK, 0, st>>>(X, xbs, xcs, xps, Cin, Lin, wp, w.b, strd, padL,
                                            Rres, alpha, beta, Out, outF32,
                                            obs, ocs, ops, ooff, Cout, Lout, tilesM);
        break;
    }
}
void convStd(hipStream_t st, const half_t* X, int Cin, int L, Conv w,
             int Kw, int strd, int padL, const half_t* Rres, float alpha, float beta,
             half_t* Out, int Cout, int Lout)
{
    convRun(st, X, Cin, L, (long)Cin * L, L, 1, w, Kw, strd, padL,
            Rres, alpha, beta, Out, (long)Cout * Lout, Lout, 1, 0, Cout, Lout);
}
void gnRun(hipStream_t st, const half_t* in, int C, int L, Gn g, int silu, half_t* out) {
    k_gn<<<Bn * NGRP, 256, 0, st>>>(in, g.g, g.b, out, C, L, NGRP, silu);
}
void bgemmRun(hipStream_t st, const half_t* A, long abss, int ars, int acs,
              const half_t* Bm, long bbs, int brs, int bcs,
              half_t* C, long cbs, int crs, int ccs,
              int M, int N, int Kd, int batch, float alpha, bool asyncOk)
{
    int tilesM = M / 16;
    dim3 g(batch * tilesM, (N + TN - 1) / TN);
    // async DMA staging requires K-contiguous tiles and exact tile coverage
    if (asyncOk && acs == 1 && brs == 1 && (M % 16) == 0 && (N % TN) == 0 && (Kd % TK) == 0)
        k_bgemm_wmma<true><<<g, BLK, 0, st>>>(A, abss, ars, acs, Bm, bbs, brs, bcs,
                                              C, cbs, crs, ccs, M, N, Kd, alpha, tilesM);
    else
        k_bgemm_wmma<false><<<g, BLK, 0, st>>>(A, abss, ars, acs, Bm, bbs, brs, bcs,
                                               C, cbs, crs, ccs, M, N, Kd, alpha, tilesM);
}
void resRun(hipStream_t st, const Res& r, const half_t* in, half_t* out,
            int Cin, int Cout, int L, half_t* tg, half_t* th, half_t* tsc)
{
    gnRun(st, in, Cin, L, r.n1, 1, tg);
    convStd(st, tg, Cin, L, r.c1, 3, 1, 1, nullptr, 1.f, 0.f, th, Cout, L);
    gnRun(st, th, Cout, L, r.n2, 1, tg);
    const half_t* res = in;
    if (r.has_sc) {
        convStd(st, in, Cin, L, r.sc, 1, 1, 0, nullptr, 1.f, 0.f, tsc, Cout, L);
        res = tsc;
    }
    convStd(st, tg, Cout, L, r.c2, 3, 1, 1, res, 1.f, 1.f, out, Cout, L);
}
void attnRun(hipStream_t st, const Attn& a, const half_t* in, half_t* out, int C, int L,
             half_t* tg, half_t* ksm, half_t* qkv, half_t* ctx, half_t* to)
{
    gnRun(st, in, C, L, a.n, 0, tg);
    convStd(st, tg, C, L, a.qkv, 1, 1, 0, nullptr, 1.f, 0.f, qkv, 3 * C, L);
    // softmax over L on the k third
    k_softmax<<<Bn * C, 128, 0, st>>>(qkv + (long)C * L, (long)3 * C * L, ksm, (long)C * L, C, L);
    // ctx[b,d,e] = sum_n k[b,d,n] * v[b,e,n]   (K-contiguous both sides -> async DMA)
    bgemmRun(st, ksm, (long)C * L, /*ars(d)*/L, /*acs(n)*/1,
             qkv + (long)2 * C * L, (long)3 * C * L, /*brs(n)*/1, /*bcs(e)*/L,
             ctx, (long)C * C, /*crs(d)*/C, /*ccs(e)*/1,
             C, C, L, Bn, 1.f, true);
    // out[b,e,n] = sum_d ctx[b,d,e] * q[b,d,n] * C^-0.5
    bgemmRun(st, ctx, (long)C * C, /*ars(e)*/1, /*acs(d)*/C,
             qkv, (long)3 * C * L, /*brs(d)*/L, /*bcs(n)*/1,
             to, (long)C * L, /*crs(e)*/L, /*ccs(n)*/1,
             C, L, C, Bn, 1.f / sqrtf((float)C), false);
    convStd(st, to, C, L, a.proj, 1, 1, 0, /*res*/in, 1.f, 1.f, out, C, L);
}
void convT2x(hipStream_t st, Conv up, const half_t* in, half_t* out, int C, int Lin, float* w2)
{
    int n = C * C;
    // even outputs: y[2l] = w3*x[l-1] + w1*x[l]
    k_pack_wT<<<(n + 255) / 256, 256, 0, st>>>(up.w, w2, C, 0);
    convRun(st, in, C, Lin, (long)C * Lin, Lin, 1, up, 2, 1, 1, nullptr, 1.f, 0.f,
            out, (long)C * 2 * Lin, 2 * Lin, 2, 0, C, Lin, w2);
    // odd outputs: y[2l+1] = w2*x[l] + w0*x[l+1]
    k_pack_wT<<<(n + 255) / 256, 256, 0, st>>>(up.w, w2, C, 1);
    convRun(st, in, C, Lin, (long)C * Lin, Lin, 1, up, 2, 1, 0, nullptr, 1.f, 0.f,
            out, (long)C * 2 * Lin, 2 * Lin, 2, 1, C, Lin, w2);
}

#define EW(krn, n, ...) krn<<<(int)(((n) + 255) / 256), 256, 0, stream>>>(__VA_ARGS__)

} // namespace

extern "C" void kernel_launch(void* const* d_in, const int* in_sizes, int n_in,
                              void* d_out, int out_size, void* d_ws, size_t ws_size,
                              hipStream_t stream)
{
    (void)in_sizes; (void)n_in; (void)out_size; (void)ws_size;

    // ---------------- parse params (JAX pytree flatten order: dict keys sorted,
    // lists in index order, None leaves dropped, each conv = (b,w), gn = (b,g))
    PIn s{d_in, 1};
    const float* x = (const float*)d_in[0];

    Conv dec_conv_in  = rdConv(s);
    Conv dec_conv_out = rdConv(s);
    Attn dec_mid_attn = rdAttn(s);
    Res  dec_mid_b1   = rdRes(s, false);
    Res  dec_mid_b2   = rdRes(s, false);
    Gn   dec_norm_out = rdGn(s);
    Res  dec_up_blocks[3][3];
    Attn dec_up2_attn[3];
    Conv dec_up1_up{nullptr, nullptr}, dec_up2_up{nullptr, nullptr};
    dec_up_blocks[0][0] = rdRes(s, true);
    dec_up_blocks[0][1] = rdRes(s, false);
    dec_up_blocks[0][2] = rdRes(s, false);
    dec_up_blocks[1][0] = rdRes(s, true);
    dec_up_blocks[1][1] = rdRes(s, false);
    dec_up_blocks[1][2] = rdRes(s, false);
    dec_up1_up = rdConv(s);
    dec_up2_attn[0] = rdAttn(s); dec_up2_attn[1] = rdAttn(s); dec_up2_attn[2] = rdAttn(s);
    dec_up_blocks[2][0] = rdRes(s, false);
    dec_up_blocks[2][1] = rdRes(s, false);
    dec_up_blocks[2][2] = rdRes(s, false);
    dec_up2_up = rdConv(s);

    Conv enc_conv_in  = rdConv(s);
    Conv enc_conv_out = rdConv(s);
    Res  enc_down_blocks[3][2];
    Attn enc_d2_attn[2];
    Conv enc_d0_down{nullptr, nullptr}, enc_d1_down{nullptr, nullptr};
    enc_down_blocks[0][0] = rdRes(s, false);
    enc_down_blocks[0][1] = rdRes(s, false);
    enc_d0_down = rdConv(s);
    enc_down_blocks[1][0] = rdRes(s, true);   // 128->256
    enc_down_blocks[1][1] = rdRes(s, false);
    enc_d1_down = rdConv(s);
    enc_d2_attn[0] = rdAttn(s); enc_d2_attn[1] = rdAttn(s);
    enc_down_blocks[2][0] = rdRes(s, true);   // 256->512
    enc_down_blocks[2][1] = rdRes(s, false);
    Attn enc_mid_attn = rdAttn(s);
    Res  enc_mid_b1   = rdRes(s, false);
    Res  enc_mid_b2   = rdRes(s, false);
    Gn   enc_norm_out = rdGn(s);

    Conv post_quant = rdConv(s);
    Conv quant      = rdConv(s);
    const float* codebook = s.next();
    Conv phi        = rdConv(s);

    // ---------------- workspace layout (activations stored f16)
    char* base = (char*)d_ws;
    size_t off = 0;
    auto alloc = [&](size_t elems, size_t esz) -> void* {
        void* p = base + off;
        off += ((elems * esz + 255) / 256) * 256;
        return p;
    };
    const size_t ACT = (size_t)Bn * 512 * 256;   // max C*L per batch in decoder
    half_t* t0   = (half_t*)alloc(ACT, 2);
    half_t* t1   = (half_t*)alloc(ACT, 2);
    half_t* tg   = (half_t*)alloc(ACT, 2);
    half_t* th   = (half_t*)alloc(ACT, 2);
    half_t* tsc  = (half_t*)alloc(ACT, 2);
    half_t* qkvb = (half_t*)alloc((size_t)Bn * 1536 * 128, 2);
    half_t* ctxb = (half_t*)alloc((size_t)Bn * 512 * 512, 2);
    const long NF = (long)Bn * 32 * 128;
    half_t* fb    = (half_t*)alloc(NF, 2);
    half_t* fhat  = (half_t*)alloc(NF, 2);
    half_t* frest = (half_t*)alloc(NF, 2);
    half_t* zb    = (half_t*)alloc(NF, 2);
    half_t* hb    = (half_t*)alloc(NF, 2);
    half_t* h2    = (half_t*)alloc(NF, 2);
    half_t* ht    = (half_t*)alloc(NF, 2);
    const long NX = (long)Bn * 512 * 66;
    half_t* xh   = (half_t*)alloc(NX, 2);
    int*    idxb = (int*)alloc(8192, 4);
    half_t* cbh  = (half_t*)alloc((size_t)8192 * 32, 2);
    float*  e2b  = (float*)alloc(8192, 4);
    float*  w2b  = (float*)alloc((size_t)512 * 512 * 2, 4);

    // =========================== ENCODER ===========================
    EW(k_f32_to_f16, NX, x, xh, NX);
    // conv_in reads x as [B, L=512, D=66] with transpose strides
    convRun(stream, xh, 66, 512, (long)512 * 66, 1, 66, enc_conv_in, 3, 1, 1,
            nullptr, 1.f, 0.f, t0, (long)128 * 512, 512, 1, 0, 128, 512);

    resRun(stream, enc_down_blocks[0][0], t0, t1, 128, 128, 512, tg, th, tsc);
    resRun(stream, enc_down_blocks[0][1], t1, t0, 128, 128, 512, tg, th, tsc);
    convStd(stream, t0, 128, 512, enc_d0_down, 3, 2, 1, nullptr, 1.f, 0.f, t1, 128, 256);
    resRun(stream, enc_down_blocks[1][0], t1, t0, 128, 256, 256, tg, th, tsc);
    resRun(stream, enc_down_blocks[1][1], t0, t1, 256, 256, 256, tg, th, tsc);
    convStd(stream, t1, 256, 256, enc_d1_down, 3, 2, 1, nullptr, 1.f, 0.f, t0, 256, 128);
    resRun(stream, enc_down_blocks[2][0], t0, t1, 256, 512, 128, tg, th, tsc);
    attnRun(stream, enc_d2_attn[0], t1, t0, 512, 128, tg, tsc, qkvb, ctxb, th);
    resRun(stream, enc_down_blocks[2][1], t0, t1, 512, 512, 128, tg, th, tsc);
    attnRun(stream, enc_d2_attn[1], t1, t0, 512, 128, tg, tsc, qkvb, ctxb, th);
    resRun(stream, enc_mid_b1, t0, t1, 512, 512, 128, tg, th, tsc);
    attnRun(stream, enc_mid_attn, t1, t0, 512, 128, tg, tsc, qkvb, ctxb, th);
    resRun(stream, enc_mid_b2, t0, t1, 512, 512, 128, tg, th, tsc);
    gnRun(stream, t1, 512, 128, enc_norm_out, 1, tg);
    convStd(stream, tg, 512, 128, enc_conv_out, 3, 1, 1, nullptr, 1.f, 0.f, ht, 32, 128);
    convStd(stream, ht, 32, 128, quant, 3, 1, 1, nullptr, 1.f, 0.f, fb, 32, 128);

    // =========================== QUANTIZE ===========================
    k_prep_cb<<<(8192 + 255) / 256, 256, 0, stream>>>(codebook, cbh, e2b, 8192, 32);
    EW(k_fill16, NF, fhat, 0.f, NF);
    EW(k_axpy16, NF, frest, fb, 1.f, (const half_t*)nullptr, 0.f, NF);
    const int pns[8] = {1, 2, 4, 8, 16, 32, 64, 128};
    for (int si_ = 0; si_ < 8; ++si_) {
        const int pn = pns[si_];
        const long nz = (long)Bn * 32 * pn;
        const half_t* zp = frest;
        if (pn != 128) {
            EW(k_pool_mean, nz, frest, zb, 32, 128, pn, nz);
            zp = zb;
        }
        k_vq_argmin_wmma<<<(Bn * pn) / 16, 32, 0, stream>>>(zp, pn, 32, cbh, e2b, 8192, idxb);
        EW(k_gather_cb, nz, codebook, idxb, hb, 32, pn, nz);
        const half_t* hup = hb;
        if (pn != 128) {
            EW(k_resize_lin, NF, hb, ht, pn, 128, NF);
            hup = ht;
        }
        // h = 0.5*h + 0.5*phi(h): fused alpha/beta in conv epilogue
        convStd(stream, hup, 32, 128, phi, 3, 1, 1, hup, 0.5f, 0.5f, h2, 32, 128);
        EW(k_axpy16, NF, fhat, fhat, 1.f, h2,  1.f, NF);
        EW(k_axpy16, NF, frest, frest, 1.f, h2, -1.f, NF);
    }
    convStd(stream, fhat, 32, 128, post_quant, 3, 1, 1, nullptr, 1.f, 0.f, ht, 32, 128);

    // =========================== DECODER ===========================
    convStd(stream, ht, 32, 128, dec_conv_in, 3, 1, 1, nullptr, 1.f, 0.f, t0, 512, 128);
    resRun(stream, dec_mid_b1, t0, t1, 512, 512, 128, tg, th, tsc);
    attnRun(stream, dec_mid_attn, t1, t0, 512, 128, tg, tsc, qkvb, ctxb, th);
    resRun(stream, dec_mid_b2, t0, t1, 512, 512, 128, tg, th, tsc);
    resRun(stream, dec_up_blocks[2][0], t1, t0, 512, 512, 128, tg, th, tsc);
    attnRun(stream, dec_up2_attn[0], t0, t1, 512, 128, tg, tsc, qkvb, ctxb, th);
    resRun(stream, dec_up_blocks[2][1], t1, t0, 512, 512, 128, tg, th, tsc);
    attnRun(stream, dec_up2_attn[1], t0, t1, 512, 128, tg, tsc, qkvb, ctxb, th);
    resRun(stream, dec_up_blocks[2][2], t1, t0, 512, 512, 128, tg, th, tsc);
    attnRun(stream, dec_up2_attn[2], t0, t1, 512, 128, tg, tsc, qkvb, ctxb, th);
    convT2x(stream, dec_up2_up, t1, t0, 512, 128, w2b);           // -> [B,512,256]
    resRun(stream, dec_up_blocks[1][0], t0, t1, 512, 256, 256, tg, th, tsc);
    resRun(stream, dec_up_blocks[1][1], t1, t0, 256, 256, 256, tg, th, tsc);
    resRun(stream, dec_up_blocks[1][2], t0, t1, 256, 256, 256, tg, th, tsc);
    convT2x(stream, dec_up1_up, t1, t0, 256, 256, w2b);           // -> [B,256,512]
    resRun(stream, dec_up_blocks[0][0], t0, t1, 256, 128, 512, tg, th, tsc);
    resRun(stream, dec_up_blocks[0][1], t1, t0, 128, 128, 512, tg, th, tsc);
    resRun(stream, dec_up_blocks[0][2], t0, t1, 128, 128, 512, tg, th, tsc);
    gnRun(stream, t1, 128, 512, dec_norm_out, 1, tg);
    convRun(stream, tg, 128, 512, (long)128 * 512, 512, 1, dec_conv_out, 3, 1, 1,
            nullptr, 1.f, 0.f, nullptr, (long)512 * 66, 1, 66, 0, 66, 512,
            nullptr, (float*)d_out);
}
